// DistillationStudentModel_40450001994011
// MI455X (gfx1250) — matrix operand order repaired and checked by reading, not compiled, not run on hardware
//
#include <hip/hip_runtime.h>
#include <hip/hip_bf16.h>
#include <math.h>

typedef __bf16 bf16_t;
typedef __attribute__((ext_vector_type(16))) __bf16 v16bf;
typedef __attribute__((ext_vector_type(8)))  __bf16 v8bf;
typedef __attribute__((ext_vector_type(4)))  __bf16 v4bf;
typedef __attribute__((ext_vector_type(8)))  float  v8f;
typedef unsigned int u32x4 __attribute__((ext_vector_type(4)));
typedef int          i32x8 __attribute__((ext_vector_type(8)));
typedef int          i32x4 __attribute__((ext_vector_type(4)));

#define B_   32
#define P_   256
#define D_   2048
#define H_   8192
#define LN_EPS 1e-5f

__device__ __forceinline__ float gelu_exact(float v) {
    return 0.5f * v * (1.0f + erff(v * 0.70710678118654752440f));
}

// ---------------------------------------------------------------------------
// Kernel 1: per-token LayerNorm + per-sample gamma/beta, bf16 out to ws.
// ---------------------------------------------------------------------------
__global__ __launch_bounds__(256)
void ln_bf16_kernel(const float* __restrict__ x,
                    const int*   __restrict__ vidx,
                    const float* __restrict__ gamma,
                    const float* __restrict__ beta,
                    bf16_t*      __restrict__ xn) {
    const int token = blockIdx.x;
    const int view  = vidx[token / P_];
    const float* xr = x  + (size_t)token * D_;
    bf16_t*      xo = xn + (size_t)token * D_;
    const float* g  = gamma + (size_t)view * D_;
    const float* bt = beta  + (size_t)view * D_;

    const int tid = threadIdx.x;
    float4 v0 = ((const float4*)xr)[tid];
    float4 v1 = ((const float4*)xr)[tid + 256];

    float s = v0.x + v0.y + v0.z + v0.w + v1.x + v1.y + v1.z + v1.w;
    float q = v0.x*v0.x + v0.y*v0.y + v0.z*v0.z + v0.w*v0.w
            + v1.x*v1.x + v1.y*v1.y + v1.z*v1.z + v1.w*v1.w;

    __shared__ float rs[256];
    __shared__ float rq[256];
    rs[tid] = s; rq[tid] = q;
    __syncthreads();
    for (int off = 128; off > 0; off >>= 1) {
        if (tid < off) { rs[tid] += rs[tid + off]; rq[tid] += rq[tid + off]; }
        __syncthreads();
    }
    const float mu   = rs[0] * (1.0f / D_);
    const float var  = rq[0] * (1.0f / D_) - mu * mu;
    const float rstd = rsqrtf(var + LN_EPS);

    float4 g0 = ((const float4*)g )[tid];
    float4 g1 = ((const float4*)g )[tid + 256];
    float4 b0 = ((const float4*)bt)[tid];
    float4 b1v= ((const float4*)bt)[tid + 256];

    v4bf o0, o1;
    o0[0] = (bf16_t)((v0.x - mu) * rstd * g0.x + b0.x);
    o0[1] = (bf16_t)((v0.y - mu) * rstd * g0.y + b0.y);
    o0[2] = (bf16_t)((v0.z - mu) * rstd * g0.z + b0.z);
    o0[3] = (bf16_t)((v0.w - mu) * rstd * g0.w + b0.w);
    o1[0] = (bf16_t)((v1.x - mu) * rstd * g1.x + b1v.x);
    o1[1] = (bf16_t)((v1.y - mu) * rstd * g1.y + b1v.y);
    o1[2] = (bf16_t)((v1.z - mu) * rstd * g1.z + b1v.z);
    o1[3] = (bf16_t)((v1.w - mu) * rstd * g1.w + b1v.w);
    ((v4bf*)xo)[tid]       = o0;
    ((v4bf*)xo)[tid + 256] = o1;
}

// ---------------------------------------------------------------------------
// GEMM building blocks.  Block tile: M=32, N=128, Kstep=64.  8 waves; wave w
// owns N columns [w*16, w*16+16) and both 16-row M sub-tiles (2 accumulators).
// A: bf16 [tokens, Kdim] staged by the Tensor Data Mover into LDS (4 KB).
// W: fp32 [Kdim, Ndim] staged coalesced, cvt to bf16, transposed [N][K] (16KB).
// Double buffered -> one barrier per K-step, 4 WMMAs/wave/K-step.
// ---------------------------------------------------------------------------

// Issue TDM: 2D tile, 32 rows x 64 bf16 elements, row pitch = row_elems,
// into contiguous LDS at lds_off.  D# layout per CDNA5 ISA.
__device__ __forceinline__ void tdm_load_a(unsigned lds_off,
                                           const bf16_t* gsrc,
                                           unsigned row_elems) {
    unsigned long long ga = (unsigned long long)(uintptr_t)(const void*)gsrc;
    u32x4 g0;
    g0[0] = 1u;                                   // count=1, user D#
    g0[1] = lds_off;                              // lds_addr (bytes)
    g0[2] = (unsigned)(ga & 0xFFFFFFFFu);         // global_addr[31:0]
    g0[3] = (unsigned)((ga >> 32) & 0x01FFFFFFu)  // global_addr[56:32]
          | 0x80000000u;                          // type=2 ("image")
    i32x8 g1;
    g1[0] = (int)(1u << 16);                      // data_size=1 (2 bytes)
    g1[1] = (int)((row_elems & 0xFFFFu) << 16);   // tensor_dim0[15:0]
    g1[2] = (int)((unsigned)(row_elems >> 16)     // tensor_dim0[31:16]
          | (((unsigned)(B_ * P_) & 0xFFFFu) << 16)); // tensor_dim1[15:0]
    g1[3] = (int)(64u << 16);                     // tile_dim0 = 64 elems
    g1[4] = 32;                                   // tile_dim1 = 32 rows
    g1[5] = (int)row_elems;                       // tensor_dim0_stride[31:0]
    g1[6] = 0;
    g1[7] = 0;
    i32x4 z4 = {0, 0, 0, 0};
    i32x8 z8 = {0, 0, 0, 0, 0, 0, 0, 0};
    // amdgpu-toolchain (clang-23) 6-arg form:
    // (g0, g1, g2, g3, g4, cpol)
    __builtin_amdgcn_tensor_load_to_lds(g0, g1, z4, z4, z8, 0);
}

__device__ __forceinline__ v16bf a_frag(const bf16_t* At, int mbase, int lane,
                                        int ks) {
    const int m  = mbase + (lane & 15);
    const int kh = lane >> 4;
    const bf16_t* row = At + m * 64 + ks;
    v8bf lo = *(const v8bf*)(row + kh * 8);
    v8bf hi = *(const v8bf*)(row + 16 + kh * 8);
    return __builtin_shufflevector(lo, hi,
        0,1,2,3,4,5,6,7,8,9,10,11,12,13,14,15);
}

__device__ __forceinline__ v16bf b_frag(const bf16_t* Wt, int wave, int lane,
                                        int ks) {
    const int n  = lane & 15;
    const int kh = lane >> 4;
    const bf16_t* col = Wt + (size_t)(wave * 16 + n) * 64 + ks;
    v8bf lo = *(const v8bf*)(col + kh * 8);
    v8bf hi = *(const v8bf*)(col + 16 + kh * 8);
    return __builtin_shufflevector(lo, hi,
        0,1,2,3,4,5,6,7,8,9,10,11,12,13,14,15);
}

// Stage 64(K) x 128(N) fp32 weights -> bf16 transposed LDS Wt[128][64].
__device__ __forceinline__ void stage_w(bf16_t* Wt, const float* Wrow0,
                                        size_t ldw, int tid,
                                        bool pf, size_t pf_off) {
    const int k    = (tid >> 4) * 4;     // 4 consecutive K rows
    const int nseg = (tid & 15) * 8;     // 8 consecutive N columns
    float rows[4][8];
#pragma unroll
    for (int i = 0; i < 4; ++i) {
        const float* r = Wrow0 + (size_t)(k + i) * ldw + nseg;
        float4 x0 = ((const float4*)r)[0];
        float4 x1 = ((const float4*)r)[1];
        rows[i][0]=x0.x; rows[i][1]=x0.y; rows[i][2]=x0.z; rows[i][3]=x0.w;
        rows[i][4]=x1.x; rows[i][5]=x1.y; rows[i][6]=x1.z; rows[i][7]=x1.w;
        if (pf) __builtin_prefetch(r + pf_off, 0, 1);   // global_prefetch_b8
    }
#pragma unroll
    for (int j = 0; j < 8; ++j) {
        v4bf p;
        p[0] = (bf16_t)rows[0][j];
        p[1] = (bf16_t)rows[1][j];
        p[2] = (bf16_t)rows[2][j];
        p[3] = (bf16_t)rows[3][j];
        *(v4bf*)(Wt + (size_t)(nseg + j) * 64 + k) = p;
    }
}

// ---------------------------------------------------------------------------
// Kernel 2: h = GELU(xn @ W1[view] + b1[view]) -> bf16 ws.  grid(BP/32, H/128)
// ---------------------------------------------------------------------------
__global__ __launch_bounds__(256)
void gemm1_gelu_kernel(const bf16_t* __restrict__ xn,
                       const float*  __restrict__ W1,
                       const float*  __restrict__ b1,
                       const int*    __restrict__ vidx,
                       bf16_t*       __restrict__ h) {
    const int token0 = blockIdx.x * 32;
    const int view   = vidx[token0 / P_];
    const int n0     = blockIdx.y * 128;
    const float* W    = W1 + (size_t)view * D_ * H_ + n0;
    const float* bias = b1 + (size_t)view * H_ + n0;
    const bf16_t* Ag  = xn + (size_t)token0 * D_;

    __shared__ __align__(32) bf16_t Abuf[2][32 * 64];
    __shared__ __align__(32) bf16_t Wbuf[2][128 * 64];

    const int tid  = threadIdx.x;
    const int lane = tid & 31;
    const int wave = tid >> 5;

    // prologue: stage tile 0
    if (wave == 0)
        tdm_load_a((unsigned)(uintptr_t)(void*)&Abuf[0][0], Ag, D_);
    stage_w(Wbuf[0], W, H_, tid, true, (size_t)64 * H_);
    if (wave == 0) __builtin_amdgcn_s_wait_tensorcnt(0);
    __syncthreads();

    v8f acc0 = {}, acc1 = {};
    const int nk = D_ / 64;
    for (int kt = 0; kt < nk; ++kt) {
        const int cur = kt & 1;
        if (kt + 1 < nk) {
            const int kb = (kt + 1) * 64;
            if (wave == 0)
                tdm_load_a((unsigned)(uintptr_t)(void*)&Abuf[cur ^ 1][0],
                           Ag + kb, D_);
            stage_w(Wbuf[cur ^ 1], W + (size_t)kb * H_, H_, tid,
                    kt + 2 < nk, (size_t)64 * H_);
        }
#pragma unroll
        for (int ks = 0; ks < 64; ks += 32) {
            v16bf a0 = a_frag(Abuf[cur], 0,  lane, ks);
            v16bf a1 = a_frag(Abuf[cur], 16, lane, ks);
            v16bf b  = b_frag(Wbuf[cur], wave, lane, ks);
            acc0 = __builtin_amdgcn_wmma_f32_16x16x32_bf16(
                       false, a0, false, b, (short)0, acc0, false, false);
            acc1 = __builtin_amdgcn_wmma_f32_16x16x32_bf16(
                       false, a1, false, b, (short)0, acc1, false, false);
        }
        if (wave == 0) __builtin_amdgcn_s_wait_tensorcnt(0);
        __syncthreads();
    }

    const int ncol  = lane & 15;
    const int khalf = lane >> 4;
    const int ng    = n0 + wave * 16 + ncol;
    const float bv  = bias[wave * 16 + ncol];
#pragma unroll
    for (int r = 0; r < 8; ++r) {
        const int m0 = r + 8 * khalf;
        h[(size_t)(token0 + m0)      * H_ + ng] = (bf16_t)gelu_exact(acc0[r] + bv);
        h[(size_t)(token0 + 16 + m0) * H_ + ng] = (bf16_t)gelu_exact(acc1[r] + bv);
    }
}

// ---------------------------------------------------------------------------
// Kernel 3: out = x + h @ W2[view] + b2[view]  (fp32).  grid(BP/32, D/128)
// ---------------------------------------------------------------------------
__global__ __launch_bounds__(256)
void gemm2_res_kernel(const bf16_t* __restrict__ h,
                      const float*  __restrict__ W2,
                      const float*  __restrict__ b2,
                      const int*    __restrict__ vidx,
                      const float*  __restrict__ x,
                      float*        __restrict__ out) {
    const int token0 = blockIdx.x * 32;
    const int view   = vidx[token0 / P_];
    const int n0     = blockIdx.y * 128;
    const float* W    = W2 + (size_t)view * H_ * D_ + n0;
    const float* bias = b2 + (size_t)view * D_ + n0;
    const bf16_t* Ag  = h + (size_t)token0 * H_;

    __shared__ __align__(32) bf16_t Abuf[2][32 * 64];
    __shared__ __align__(32) bf16_t Wbuf[2][128 * 64];

    const int tid  = threadIdx.x;
    const int lane = tid & 31;
    const int wave = tid >> 5;

    if (wave == 0)
        tdm_load_a((unsigned)(uintptr_t)(void*)&Abuf[0][0], Ag, H_);
    stage_w(Wbuf[0], W, D_, tid, true, (size_t)64 * D_);
    if (wave == 0) __builtin_amdgcn_s_wait_tensorcnt(0);
    __syncthreads();

    v8f acc0 = {}, acc1 = {};
    const int nk = H_ / 64;
    for (int kt = 0; kt < nk; ++kt) {
        const int cur = kt & 1;
        if (kt + 1 < nk) {
            const int kb = (kt + 1) * 64;
            if (wave == 0)
                tdm_load_a((unsigned)(uintptr_t)(void*)&Abuf[cur ^ 1][0],
                           Ag + kb, H_);
            stage_w(Wbuf[cur ^ 1], W + (size_t)kb * D_, D_, tid,
                    kt + 2 < nk, (size_t)64 * D_);
        }
#pragma unroll
        for (int ks = 0; ks < 64; ks += 32) {
            v16bf a0 = a_frag(Abuf[cur], 0,  lane, ks);
            v16bf a1 = a_frag(Abuf[cur], 16, lane, ks);
            v16bf b  = b_frag(Wbuf[cur], wave, lane, ks);
            acc0 = __builtin_amdgcn_wmma_f32_16x16x32_bf16(
                       false, a0, false, b, (short)0, acc0, false, false);
            acc1 = __builtin_amdgcn_wmma_f32_16x16x32_bf16(
                       false, a1, false, b, (short)0, acc1, false, false);
        }
        if (wave == 0) __builtin_amdgcn_s_wait_tensorcnt(0);
        __syncthreads();
    }

    const int ncol  = lane & 15;
    const int khalf = lane >> 4;
    const int ng    = n0 + wave * 16 + ncol;
    const float bv  = bias[wave * 16 + ncol];
#pragma unroll
    for (int r = 0; r < 8; ++r) {
        const int m0 = r + 8 * khalf;
        const size_t o0 = (size_t)(token0 + m0)      * D_ + ng;
        const size_t o1 = (size_t)(token0 + 16 + m0) * D_ + ng;
        out[o0] = x[o0] + acc0[r] + bv;
        out[o1] = x[o1] + acc1[r] + bv;
    }
}

// ---------------------------------------------------------------------------
extern "C" void kernel_launch(void* const* d_in, const int* in_sizes, int n_in,
                              void* d_out, int out_size, void* d_ws, size_t ws_size,
                              hipStream_t stream) {
    const float* x     = (const float*)d_in[0];   // [B,P,D]
    const int*   vidx  = (const int*)  d_in[1];   // [B]
    const float* gamma = (const float*)d_in[2];   // [V,D]
    const float* beta  = (const float*)d_in[3];   // [V,D]
    const float* W1    = (const float*)d_in[4];   // [V,D,H]
    const float* b1    = (const float*)d_in[5];   // [V,H]
    const float* W2    = (const float*)d_in[6];   // [V,H,D]
    const float* b2    = (const float*)d_in[7];   // [V,D]
    float* out = (float*)d_out;

    bf16_t* xn = (bf16_t*)d_ws;
    bf16_t* h  = (bf16_t*)((char*)d_ws + (size_t)B_ * P_ * D_ * sizeof(bf16_t));

    ln_bf16_kernel<<<B_ * P_, 256, 0, stream>>>(x, vidx, gamma, beta, xn);
    gemm1_gelu_kernel<<<dim3(B_ * P_ / 32, H_ / 128), 256, 0, stream>>>(
        xn, W1, b1, vidx, h);
    gemm2_res_kernel<<<dim3(B_ * P_ / 32, D_ / 128), 256, 0, stream>>>(
        h, W2, b2, vidx, x, out);
}
